// VanillaRNN_52647709115018
// MI455X (gfx1250) — compile-verified, matrix operand order checked by reading, not run on hardware
//
#include <hip/hip_runtime.h>
#include <hip/hip_bf16.h>
#include <math.h>

typedef __attribute__((ext_vector_type(16))) __bf16 v16bf;
typedef __attribute__((ext_vector_type(8)))  float  v8f;

#define SEQ  1024
#define HID  512
#define ODIM 10
#define MT   16          // rows per workgroup
#define NKT  (HID / 32)  // 16 k-tiles of 32
#define NNT  (HID / 16)  // 32 n-tiles of 16

__device__ __forceinline__ __bf16 f2bf(float f) { return (__bf16)f; }

// Branch-free tanh: one v_exp_f32 + one v_rcp_f32, no EXEC divergence.
// Accuracy ~1e-7 relative for |x| small, exact saturation for large |x| —
// far beyond the bf16 precision the recurrence actually carries.
__device__ __forceinline__ float fast_tanh(float x) {
    float a = fabsf(x);
    float e = __expf(-2.0f * a);                      // v_exp_f32
    float t = (1.0f - e) * __builtin_amdgcn_rcpf(1.0f + e); // v_rcp_f32
    return copysignf(t, x);
}

// ---------------------------------------------------------------------------
// Pre-pass: swizzle W_hh (f32, [N=512][K=512] row-major, B[k][n] = W_hh[n][k])
// into bf16 fragments matching the CDNA5 16-bit B-matrix VGPR layout:
//   frag index = kt*32 + nt ; per lane (n = lane%16, khalf = lane/16),
//   element e -> k32 = (e/8)*16 + khalf*8 + ((e/2)%4)*2 + (e%2)
// Each lane's 16 bf16 elements are stored contiguously (32 bytes).
// ---------------------------------------------------------------------------
__global__ void swizzle_whh_kernel(const float* __restrict__ Whh,
                                   __bf16* __restrict__ wsB) {
    int unit = blockIdx.x * (blockDim.x >> 5) + (threadIdx.x >> 5); // 0..511
    int lane = threadIdx.x & 31;
    int kt = unit >> 5;        // 0..15
    int nt = unit & 31;        // 0..31
    int nloc = lane & 15;
    int kh   = lane >> 4;
    int ng = nt * 16 + nloc;
    __bf16* out = wsB + (((size_t)unit * 32) + lane) * 16;
#pragma unroll
    for (int e = 0; e < 16; ++e) {
        int v = e >> 1, p = e & 1;
        int k32 = (v >> 2) * 16 + kh * 8 + (v & 3) * 2 + p;
        int kg = kt * 32 + k32;
        out[e] = f2bf(Whh[(size_t)ng * HID + kg]);
    }
}

// ---------------------------------------------------------------------------
// Persistent recurrence kernel: 32 workgroups x 512 threads (16 wave32s).
// WG w owns rows [16w, 16w+16). Each wave owns n-tiles {2*wave, 2*wave+1}.
// Per step: D[16x512] = H[16x512] @ W_hh^T via chained v_wmma_f32_16x16x32_bf16,
// branch-free tanh epilogue, bf16 A-fragments written to double-buffered LDS.
// ---------------------------------------------------------------------------
__global__ void __launch_bounds__(512, 1)
rnn_wmma_kernel(const float* __restrict__ x,
                const float* __restrict__ Whx,
                const float* __restrict__ bhx,
                const float* __restrict__ bhh,
                const float* __restrict__ Woh,
                const float* __restrict__ boh,
                const __bf16* __restrict__ wsB,
                float* __restrict__ out) {
    __shared__ __align__(32) __bf16 ha[2][NKT][32][16]; // A-fragment layout, 2x16KB
    __shared__ float hf[MT][HID];                        // final H (f32) for output GEMV
    __shared__ float lx[2][MT];                          // x[:, t] slice, double buffered
    __shared__ float os[MT][ODIM];

    const int tid   = threadIdx.x;
    const int lane  = tid & 31;
    const int wave  = tid >> 5;
    const int rowb  = blockIdx.x * MT;

    const int nt0 = wave * 2, nt1 = wave * 2 + 1;
    const int nloc  = lane & 15;
    const int lhalf = lane >> 4;
    const int ng0 = nt0 * 16 + nloc;
    const int ng1 = nt1 * 16 + nloc;

    // Per-lane input-projection constants for this lane's two output columns.
    const float whx0 = Whx[ng0], whx1 = Whx[ng1];
    const float bs0 = bhx[ng0] + bhh[ng0];
    const float bs1 = bhx[ng1] + bhh[ng1];

    // A-fragment target coordinates for k = ng (this lane's D columns feed
    // next step's K dimension): k -> (k_tile, khalf, element) — constant per lane.
    auto amap = [](int k, int& kt2, int& kh, int& e) {
        kt2 = k >> 5;
        int k32 = k & 31;
        int group = k32 >> 4, rem = k32 & 15;
        kh = rem >> 3;
        e = group * 8 + (rem & 7);
    };
    int kt2_0, kh0, e0, kt2_1, kh1, e1;
    amap(ng0, kt2_0, kh0, e0);
    amap(ng1, kt2_1, kh1, e1);

    // h0 = 0: zero A-fragment buffer 0; preload x[:, 0].
    for (int i = tid; i < NKT * 32 * 16; i += blockDim.x)
        ((__bf16*)ha[0])[i] = f2bf(0.0f);
    if (tid < MT) lx[0][tid] = x[(size_t)(rowb + tid) * SEQ];
    __syncthreads();

    for (int t = 0; t < SEQ; ++t) {
        const int rbuf = t & 1;
        const int wbuf = rbuf ^ 1;

        v8f c0 = {};
        v8f c1 = {};
#pragma unroll 4
        for (int kt = 0; kt < NKT; ++kt) {
            v16bf a = *(const v16bf*)&ha[rbuf][kt][lane][0];
            const __bf16* bp0 = wsB + ((((size_t)kt * 32) + nt0) * 32 + lane) * 16;
            const __bf16* bp1 = wsB + ((((size_t)kt * 32) + nt1) * 32 + lane) * 16;
            v16bf b0 = *(const v16bf*)bp0;
            v16bf b1 = *(const v16bf*)bp1;
            __builtin_prefetch(bp0 + (size_t)32 * 32 * 16, 0, 1); // next k-tile
            c0 = __builtin_amdgcn_wmma_f32_16x16x32_bf16(false, a, false, b0,
                                                         (short)0, c0, false, false);
            c1 = __builtin_amdgcn_wmma_f32_16x16x32_bf16(false, a, false, b1,
                                                         (short)0, c1, false, false);
        }

        // Epilogue: D lane holds N = lane%16, VGPR r holds M = r + 8*(lane/16).
        float v0s[8], v1s[8];
#pragma unroll
        for (int r = 0; r < 8; ++r) {
            int m = r + (lhalf << 3);
            float xv = lx[rbuf][m];
            float v0 = fast_tanh(c0[r] + xv * whx0 + bs0);
            float v1 = fast_tanh(c1[r] + xv * whx1 + bs1);
            v0s[r] = v0;
            v1s[r] = v1;
            ha[wbuf][kt2_0][(kh0 << 4) + m][e0] = f2bf(v0);
            ha[wbuf][kt2_1][(kh1 << 4) + m][e1] = f2bf(v1);
        }
        if (t + 1 < SEQ) {
            if (tid < MT)
                lx[wbuf][tid] = x[(size_t)(rowb + tid) * SEQ + (t + 1)];
        } else {
            // Final step: spill full-precision H for the output projection.
#pragma unroll
            for (int r = 0; r < 8; ++r) {
                int m = r + (lhalf << 3);
                hf[m][ng0] = v0s[r];
                hf[m][ng1] = v1s[r];
            }
        }
        __syncthreads();
    }

    // Output projection: o[16][10] = hf @ W_oh^T + b_oh
    if (tid < MT * ODIM) {
        int b = tid / ODIM, oj = tid % ODIM;
        float acc = boh[oj];
        const float* wr = Woh + (size_t)oj * HID;
        for (int j = 0; j < HID; ++j) acc += hf[b][j] * wr[j];
        os[b][oj] = acc;
    }
    __syncthreads();

    // Softmax per row, one thread per row.
    if (tid < MT) {
        float mx = os[tid][0];
        for (int j = 1; j < ODIM; ++j) mx = fmaxf(mx, os[tid][j]);
        float e[ODIM], s = 0.0f;
        for (int j = 0; j < ODIM; ++j) { e[j] = __expf(os[tid][j] - mx); s += e[j]; }
        float inv = 1.0f / s;
        float* orow = out + (size_t)(rowb + tid) * ODIM;
        for (int j = 0; j < ODIM; ++j) orow[j] = e[j] * inv;
    }
}

extern "C" void kernel_launch(void* const* d_in, const int* in_sizes, int n_in,
                              void* d_out, int out_size, void* d_ws, size_t ws_size,
                              hipStream_t stream) {
    const float* x   = (const float*)d_in[0];
    const float* Whx = (const float*)d_in[1];
    const float* bhx = (const float*)d_in[2];
    const float* Whh = (const float*)d_in[3];
    const float* bhh = (const float*)d_in[4];
    const float* Woh = (const float*)d_in[5];
    const float* boh = (const float*)d_in[6];
    __bf16* wsB = (__bf16*)d_ws;   // 512*512 bf16 = 512 KB swizzled W_hh

    // 512 fragment units, 4 per block of 128 threads.
    swizzle_whh_kernel<<<128, 128, 0, stream>>>(Whh, wsB);
    // 32 row-blocks, fully independent across the whole recurrence.
    rnn_wmma_kernel<<<32, 512, 0, stream>>>(x, Whx, bhx, bhh, Woh, boh, wsB,
                                            (float*)d_out);
}
